// GATGraphClassifier_52020643889634
// MI455X (gfx1250) — matrix operand order repaired and checked by reading, not compiled, not run on hardware
//
#include <hip/hip_runtime.h>
#include <stdint.h>
#include <stddef.h>

// ---------- types for WMMA ----------
typedef __bf16 bf16_t;
typedef __attribute__((ext_vector_type(16))) __bf16 v16bf;
typedef __attribute__((ext_vector_type(8)))  __bf16 v8bf;
typedef __attribute__((ext_vector_type(8)))  float  v8f;

#define NG   256      // graphs
#define NPG0 256      // nodes per graph, layer-1 input
#define NE   262144   // edges (fixed, masked)
#define NH   3        // heads
#define NC   128      // channels per head
#define HC   384      // NH*NC
#define PADK 8        // LDS row padding (halves) to break bank conflicts

// ---------------------------------------------------------------- utilities
__device__ __forceinline__ float leaky02(float v) { return v > 0.f ? v : 0.2f * v; }

__device__ __forceinline__ void atomicMaxF(float* addr, float val) {
  int* ai = (int*)addr;
  int old = __float_as_int(*addr);
  while (__int_as_float(old) < val) {
    int assumed = old;
    old = atomicCAS(ai, assumed, __float_as_int(val));
    if (old == assumed) break;
  }
}

__global__ void k_fill_zero(float* p, int n) {
  int i = blockIdx.x * blockDim.x + threadIdx.x;
  if (i < n) p[i] = 0.f;
}

__global__ void k_cast_bf16(const float* __restrict__ in, bf16_t* __restrict__ out, int n) {
  int i = blockIdx.x * blockDim.x + threadIdx.x;
  if (i < n) out[i] = (bf16_t)in[i];
}

// out[c*rows + r] = in[r*cols + c]  (f32 -> bf16 transpose, for weight B^T)
__global__ void k_transpose_cast(const float* __restrict__ in, bf16_t* __restrict__ out,
                                 int rows, int cols) {
  int i = blockIdx.x * blockDim.x + threadIdx.x;
  if (i >= rows * cols) return;
  int r = i / cols, c = i - r * cols;
  out[c * rows + r] = (bf16_t)in[i];
}

// out = bf16(in + bias[col])   (row width = HC)
__global__ void k_bias_cast(const float* __restrict__ in, const float* __restrict__ bias,
                            bf16_t* __restrict__ out, int n) {
  int i = blockIdx.x * blockDim.x + threadIdx.x;
  if (i >= n) return;
  int col = i % HC;
  out[i] = (bf16_t)(in[i] + bias[col]);
}

// ---------------------------------------------------------------- WMMA GEMM
// C[M,N] = A[M,K](bf16 row-major) * Bt[N,K](bf16, B pre-transposed) (+ bias[N]).
// Block = 256 thr (8 waves) -> 128 rows x 64 cols; grid = (M/128, N/64).
// B stripe [64][K] staged in LDS (padded rows). Each wave: 16x64 tile,
// 4 f32 accumulators; K unrolled by 64 with ping-pong A fragments (no reg
// copies -> no WMMA->VALU hazard nops); all 4 B fragments loaded before the
// WMMA group so dscnt waits can be partial.
__global__ void __launch_bounds__(256)
k_gemm_bf16(const bf16_t* __restrict__ A, const bf16_t* __restrict__ Bt,
            const float* __restrict__ bias, float* __restrict__ C,
            int M, int N, int K) {
  __shared__ bf16_t bs[64 * (HC + PADK)];   // up to 64 x (384+8) halves = 49 KB
  const int tid  = threadIdx.x;
  const int lane = tid & 31;
  const int wv   = tid >> 5;
  const int col0 = blockIdx.y << 6;
  const int ldb  = K + PADK;

  const int row0 = (blockIdx.x << 7) + (wv << 4);
  const int hi   = lane >> 4;                 // 0: lanes 0-15, 1: lanes 16-31
  const bf16_t* ap = A + (size_t)(row0 + (lane & 15)) * K + (hi << 3);
  // pull this wave's A rows toward the caches while we stage B (gfx1250 prefetch)
  if (row0 < M) __builtin_prefetch(ap, 0, 0);

  // ---- stage B stripe [64][K] -> LDS, rows padded by PADK halves ----
  const int cpr = K >> 3;                     // v8bf chunks per row
  for (int i = tid; i < 64 * cpr; i += 256) {
    int r = i / cpr, c = i - r * cpr;
    *(v8bf*)(bs + r * ldb + (c << 3)) =
        *(const v8bf*)(Bt + (size_t)(col0 + r) * K + (c << 3));
  }
  __syncthreads();
  if (row0 >= M) return;

  // B fragment base: column nlocal = 16*j + (lane&15), K offset 8*hi
  const bf16_t* bsl = bs + (lane & 15) * ldb + (hi << 3);

  union AB { v16bf v; v8bf h[2]; };
  // A fragment (ISA 16-bit A 16x32 layout): halves 0..7 = K k0+8*hi..,
  // halves 8..15 = K k0+16+8*hi..
  AB a0, a1;
  a0.h[0] = *(const v8bf*)(ap);
  a0.h[1] = *(const v8bf*)(ap + 16);
  a1.h[0] = *(const v8bf*)(ap + 32);
  a1.h[1] = *(const v8bf*)(ap + 48);

  v8f acc[4] = {v8f{}, v8f{}, v8f{}, v8f{}};
  for (int k0 = 0; k0 < K; k0 += 64) {        // K % 64 == 0 (128 or 384)
    AB b[4];
#pragma unroll
    for (int j = 0; j < 4; ++j) {             // all 8 LDS loads first
      const bf16_t* bp = bsl + j * 16 * ldb + k0;
      b[j].h[0] = *(const v8bf*)(bp);
      b[j].h[1] = *(const v8bf*)(bp + 16);
    }
#pragma unroll
    for (int j = 0; j < 4; ++j)
      acc[j] = __builtin_amdgcn_wmma_f32_16x16x32_bf16(
                   false, a0.v, false, b[j].v, (short)0, acc[j], false, false);
    if (k0 + 64 < K) {                        // stream next A (even slot)
      a0.h[0] = *(const v8bf*)(ap + k0 + 64);
      a0.h[1] = *(const v8bf*)(ap + k0 + 80);
    }
#pragma unroll
    for (int j = 0; j < 4; ++j) {
      const bf16_t* bp = bsl + j * 16 * ldb + k0 + 32;
      b[j].h[0] = *(const v8bf*)(bp);
      b[j].h[1] = *(const v8bf*)(bp + 16);
    }
#pragma unroll
    for (int j = 0; j < 4; ++j)
      acc[j] = __builtin_amdgcn_wmma_f32_16x16x32_bf16(
                   false, a1.v, false, b[j].v, (short)0, acc[j], false, false);
    if (k0 + 96 < K) {                        // stream next A (odd slot)
      a1.h[0] = *(const v8bf*)(ap + k0 + 96);
      a1.h[1] = *(const v8bf*)(ap + k0 + 112);
    }
  }

  // C/D layout: element i -> row row0 + i + 8*hi, col = col0 + 16*j + (lane&15)
  const int mr = row0 + (hi << 3);
  const int nc = lane & 15;
#pragma unroll
  for (int j = 0; j < 4; ++j) {
    const int n = col0 + 16 * j + nc;
    const float bv = bias ? bias[n] : 0.f;
#pragma unroll
    for (int i = 0; i < 8; ++i)
      C[(size_t)(mr + i) * N + n] = acc[j][i] + bv;
  }
}

// ---------------------------------------------------------------- GAT attention
__global__ void k_attn_logits(const float* __restrict__ h, const float* __restrict__ as,
                              const float* __restrict__ ad, float* __restrict__ als,
                              float* __restrict__ ald, int n) {
  int idx = blockIdx.x * blockDim.x + threadIdx.x;
  if (idx >= n * NH) return;
  int node = idx / NH, hh = idx - node * NH;
  const float* hp = h + (size_t)node * HC + hh * NC;
  const float* ap = as + hh * NC;
  const float* dp = ad + hh * NC;
  float s = 0.f, d = 0.f;
  for (int c = 0; c < NC; ++c) { float v = hp[c]; s += v * ap[c]; d += v * dp[c]; }
  als[idx] = s; ald[idx] = d;
}

__global__ void k_self_init(const float* __restrict__ als, const float* __restrict__ ald,
                            float* __restrict__ amax, int n3) {
  int i = blockIdx.x * blockDim.x + threadIdx.x;
  if (i < n3) amax[i] = leaky02(als[i] + ald[i]);
}

__global__ void k_edge_max(const float* __restrict__ als, const float* __restrict__ ald,
                           const int* __restrict__ src, const int* __restrict__ dst,
                           const int* __restrict__ mask, float* __restrict__ amax, int ne3) {
  int i = blockIdx.x * blockDim.x + threadIdx.x;
  if (i >= ne3) return;
  int e = i / NH, hh = i - e * NH;
  if (!mask[e]) return;
  int s = src[e], d = dst[e];
  float v = leaky02(als[s * NH + hh] + ald[d * NH + hh]);
  atomicMaxF(&amax[d * NH + hh], v);
}

__global__ void k_den_self(const float* __restrict__ als, const float* __restrict__ ald,
                           const float* __restrict__ amax, float* __restrict__ den, int n3) {
  int i = blockIdx.x * blockDim.x + threadIdx.x;
  if (i < n3) den[i] = __expf(leaky02(als[i] + ald[i]) - amax[i]);
}

__global__ void k_edge_den(const float* __restrict__ als, const float* __restrict__ ald,
                           const int* __restrict__ src, const int* __restrict__ dst,
                           const int* __restrict__ mask, const float* __restrict__ amax,
                           float* __restrict__ exw, float* __restrict__ den, int ne3) {
  int i = blockIdx.x * blockDim.x + threadIdx.x;
  if (i >= ne3) return;
  int e = i / NH, hh = i - e * NH;
  if (!mask[e]) { exw[i] = 0.f; return; }
  int s = src[e], d = dst[e];
  float ex = __expf(leaky02(als[s * NH + hh] + ald[d * NH + hh]) - amax[d * NH + hh]);
  exw[i] = ex;
  atomicAdd(&den[d * NH + hh], ex);
}

__global__ void k_edge_w(const int* __restrict__ dst, const int* __restrict__ mask,
                         const float* __restrict__ den, float* __restrict__ exw, int ne3) {
  int i = blockIdx.x * blockDim.x + threadIdx.x;
  if (i >= ne3) return;
  int e = i / NH, hh = i - e * NH;
  if (mask[e]) exw[i] /= den[dst[e] * NH + hh];
}

__global__ void k_wself(const float* __restrict__ als, const float* __restrict__ ald,
                        const float* __restrict__ amax, const float* __restrict__ den,
                        float* __restrict__ wself, int n3) {
  int i = blockIdx.x * blockDim.x + threadIdx.x;
  if (i < n3) wself[i] = __expf(leaky02(als[i] + ald[i]) - amax[i]) / den[i];
}

__global__ void k_agg_self(const float* __restrict__ h, const float* __restrict__ wself,
                           float* __restrict__ out, int nHC) {
  int i = blockIdx.x * blockDim.x + threadIdx.x;
  if (i >= nHC) return;
  int node = i / HC, col = i - node * HC;
  out[i] = wself[node * NH + (col >> 7)] * h[i];
}

__global__ void k_edge_agg(const float* __restrict__ hmat, const float* __restrict__ w,
                           const int* __restrict__ src, const int* __restrict__ dst,
                           const int* __restrict__ mask, float* __restrict__ out) {
  int e = blockIdx.x;
  if (!mask[e]) return;
  int s = src[e], d = dst[e], tid = threadIdx.x;
#pragma unroll
  for (int hh = 0; hh < NH; ++hh) {
    float wv = w[e * NH + hh];
    atomicAdd(out + (size_t)d * HC + hh * NC + tid,
              wv * hmat[(size_t)s * HC + hh * NC + tid]);
  }
}

// ---------------------------------------------------------------- edges / pooling
__global__ void k_copy_edges(const int* __restrict__ ei, int* __restrict__ src,
                             int* __restrict__ dst, int* __restrict__ mask, int ne) {
  int e = blockIdx.x * blockDim.x + threadIdx.x;
  if (e >= ne) return;
  src[e] = ei[e]; dst[e] = ei[ne + e]; mask[e] = 1;
}

__global__ void k_topk(const float* __restrict__ x, const float* __restrict__ p,
                       float* __restrict__ xout, int* __restrict__ newid,
                       int npg, int k) {
  __shared__ float sc[256];
  __shared__ float pnorm;
  __shared__ float red[128];
  int g = blockIdx.x, tid = threadIdx.x;
  if (tid < 128) red[tid] = p[tid] * p[tid];
  __syncthreads();
  if (tid == 0) { float s = 0.f; for (int i = 0; i < 128; ++i) s += red[i]; pnorm = sqrtf(s); }
  __syncthreads();
  float inv = 1.0f / pnorm;
  int node = g * npg + tid;
  float score = -1e30f;
  if (tid < npg) {
    const float* xp = x + (size_t)node * NC;
    float dot = 0.f;
    for (int c = 0; c < NC; ++c) dot += xp[c] * p[c];
    score = tanhf(dot * inv);
  }
  sc[tid] = score;
  __syncthreads();
  if (tid < npg) {
    int rank = 0;
    for (int j = 0; j < npg; ++j) {
      float o = sc[j];
      rank += (o > score) || (o == score && j < tid);
    }
    if (rank < k) {
      int nid = g * k + rank;
      newid[node] = nid;
      const float* xp = x + (size_t)node * NC;
      float* xo = xout + (size_t)nid * NC;
      for (int c = 0; c < NC; ++c) xo[c] = xp[c] * score;
    } else {
      newid[node] = -1;
    }
  }
}

__global__ void k_remap(int* __restrict__ src, int* __restrict__ dst, int* __restrict__ mask,
                        const int* __restrict__ newid, int ne) {
  int e = blockIdx.x * blockDim.x + threadIdx.x;
  if (e >= ne) return;
  int ns = -1, nd = -1;
  if (mask[e]) { ns = newid[src[e]]; nd = newid[dst[e]]; }
  mask[e] = (ns >= 0 && nd >= 0);
  src[e] = ns > 0 ? ns : 0;
  dst[e] = nd > 0 ? nd : 0;
}

__global__ void k_readout(const float* __restrict__ x, float* __restrict__ xsum, int k) {
  int g = blockIdx.x, c = threadIdx.x;   // 128 threads
  const float* xp = x + (size_t)g * k * NC + c;
  float m = -1e30f, s = 0.f;
  for (int i = 0; i < k; ++i) { float v = xp[(size_t)i * NC]; m = fmaxf(m, v); s += v; }
  xsum[g * 256 + c] += m;
  xsum[g * 256 + 128 + c] += s / (float)k;
}

// ---------------------------------------------------------------- MLP head
__global__ void k_mlp(const float* __restrict__ xsum, const float* __restrict__ w1,
                      const float* __restrict__ b1, const float* __restrict__ w2,
                      const float* __restrict__ b2, float* __restrict__ out) {
  __shared__ float xr[256];
  __shared__ float hb[128];
  __shared__ float lg[10];
  int g = blockIdx.x, tid = threadIdx.x;   // 128 threads
  xr[tid]       = xsum[g * 256 + tid];
  xr[tid + 128] = xsum[g * 256 + 128 + tid];
  __syncthreads();
  float acc = b1[tid];
  for (int i = 0; i < 256; ++i) acc += xr[i] * w1[i * 128 + tid];
  hb[tid] = fmaxf(acc, 0.f);
  __syncthreads();
  if (tid < 10) {
    float a = b2[tid];
    for (int i = 0; i < 128; ++i) a += hb[i] * w2[i * 10 + tid];
    lg[tid] = a;
  }
  __syncthreads();
  if (tid < 10) {
    float mx = lg[0];
    for (int i = 1; i < 10; ++i) mx = fmaxf(mx, lg[i]);
    float s = 0.f;
    for (int i = 0; i < 10; ++i) s += expf(lg[i] - mx);
    out[g * 10 + tid] = lg[tid] - (mx + logf(s));
  }
}

// ---------------------------------------------------------------- host driver
static inline int cdiv(long a, long b) { return (int)((a + b - 1) / b); }

extern "C" void kernel_launch(void* const* d_in, const int* in_sizes, int n_in,
                              void* d_out, int out_size, void* d_ws, size_t ws_size,
                              hipStream_t stream) {
  (void)in_sizes; (void)n_in; (void)out_size; (void)ws_size;
  const float* x0   = (const float*)d_in[0];
  const int*   eidx = (const int*)d_in[1];
  const float* l1w  = (const float*)d_in[24];
  const float* l1b  = (const float*)d_in[25];
  const float* l2w  = (const float*)d_in[26];
  const float* l2b  = (const float*)d_in[27];

  const int NMAX = NG * NPG0;  // 65536
  size_t off = 0;
  auto alloc = [&](size_t bytes) { size_t o = off; off += (bytes + 255) & ~(size_t)255; return o; };
  char* W = (char*)d_ws;
  bf16_t* xb    = (bf16_t*)(W + alloc((size_t)NMAX * HC * 2)); // bf16 input / gat_out
  float*  hbuf  = (float*)(W + alloc((size_t)NMAX * HC * 4));  // h = x@W
  float*  gat   = (float*)(W + alloc((size_t)NMAX * HC * 4));  // aggregated
  float*  als   = (float*)(W + alloc((size_t)NMAX * NH * 4));
  float*  ald   = (float*)(W + alloc((size_t)NMAX * NH * 4));
  float*  amax  = (float*)(W + alloc((size_t)NMAX * NH * 4));
  float*  den   = (float*)(W + alloc((size_t)NMAX * NH * 4));
  float*  wself = (float*)(W + alloc((size_t)NMAX * NH * 4));
  float*  exw   = (float*)(W + alloc((size_t)NE * NH * 4));
  float*  xA    = (float*)(W + alloc((size_t)NMAX * NC * 4));  // GEMM2 out
  float*  xB    = (float*)(W + alloc((size_t)NMAX * NC * 4));  // pooled out
  int*    newid = (int*)(W + alloc((size_t)NMAX * 4));
  int*    srcb  = (int*)(W + alloc((size_t)NE * 4));
  int*    dstb  = (int*)(W + alloc((size_t)NE * 4));
  int*    maskb = (int*)(W + alloc((size_t)NE * 4));
  bf16_t* wt    = (bf16_t*)(W + alloc((size_t)128 * HC * 2));  // transposed weight buf
  float*  xsum  = (float*)(W + alloc((size_t)NG * 256 * 4));

  const int widx[3] = {3, 10, 17};
  const int KK[3]   = {205, 164, 132};

  k_fill_zero<<<cdiv(NG * 256, 256), 256, 0, stream>>>(xsum, NG * 256);
  k_copy_edges<<<cdiv(NE, 256), 256, 0, stream>>>(eidx, srcb, dstb, maskb, NE);

  int n = NMAX, npg = NPG0;
  const float* xin = x0;
  for (int L = 0; L < 3; ++L) {
    const float* Wl = (const float*)d_in[widx[L] + 0];
    const float* as = (const float*)d_in[widx[L] + 1];
    const float* ad = (const float*)d_in[widx[L] + 2];
    const float* bl = (const float*)d_in[widx[L] + 3];
    const float* tw = (const float*)d_in[widx[L] + 4];
    const float* tb = (const float*)d_in[widx[L] + 5];
    const float* pl = (const float*)d_in[widx[L] + 6];
    const int n3 = n * NH, ne3 = NE * NH, nHC = n * HC;

    // ---- GEMM1: h[n,384] = bf16(x) @ W  (Wt: [384,128]) ----
    k_transpose_cast<<<cdiv(128 * HC, 256), 256, 0, stream>>>(Wl, wt, 128, HC);
    k_cast_bf16<<<cdiv((long)n * NC, 256), 256, 0, stream>>>(xin, xb, n * NC);
    {
      dim3 g(n / 128, HC / 64);
      k_gemm_bf16<<<g, 256, 0, stream>>>(xb, wt, nullptr, hbuf, n, HC, 128);
    }
    // ---- attention softmax over incoming edges (+ self loop) ----
    k_attn_logits<<<cdiv(n3, 256), 256, 0, stream>>>(hbuf, as, ad, als, ald, n);
    k_self_init<<<cdiv(n3, 256), 256, 0, stream>>>(als, ald, amax, n3);
    k_edge_max<<<cdiv(ne3, 256), 256, 0, stream>>>(als, ald, srcb, dstb, maskb, amax, ne3);
    k_den_self<<<cdiv(n3, 256), 256, 0, stream>>>(als, ald, amax, den, n3);
    k_edge_den<<<cdiv(ne3, 256), 256, 0, stream>>>(als, ald, srcb, dstb, maskb, amax, exw, den, ne3);
    k_edge_w<<<cdiv(ne3, 256), 256, 0, stream>>>(dstb, maskb, den, exw, ne3);
    k_wself<<<cdiv(n3, 256), 256, 0, stream>>>(als, ald, amax, den, wself, n3);
    k_agg_self<<<cdiv((long)nHC, 256), 256, 0, stream>>>(hbuf, wself, gat, nHC);
    k_edge_agg<<<NE, 128, 0, stream>>>(hbuf, exw, srcb, dstb, maskb, gat);
    // ---- GEMM2: xA[n,128] = bf16(gat + b) @ tw + tb  (twt: [128,384]) ----
    k_bias_cast<<<cdiv((long)nHC, 256), 256, 0, stream>>>(gat, bl, xb, nHC);
    k_transpose_cast<<<cdiv(HC * 128, 256), 256, 0, stream>>>(tw, wt, HC, 128);
    {
      dim3 g(n / 128, NC / 64);
      k_gemm_bf16<<<g, 256, 0, stream>>>(xb, wt, tb, xA, n, NC, HC);
    }
    // ---- TopK pooling + edge remap + readout ----
    int k = KK[L];
    k_topk<<<NG, 256, 0, stream>>>(xA, pl, xB, newid, npg, k);
    k_remap<<<cdiv(NE, 256), 256, 0, stream>>>(srcb, dstb, maskb, newid, NE);
    k_readout<<<NG, 128, 0, stream>>>(xB, xsum, k);
    n = NG * k; npg = k; xin = xB;
  }

  k_mlp<<<NG, 128, 0, stream>>>(xsum, l1w, l1b, l2w, l2b, (float*)d_out);
}
